// Attention_72310069396149
// MI455X (gfx1250) — matrix operand order repaired and checked by reading, not compiled, not run on hardware
//
#include <hip/hip_runtime.h>
#include <hip/hip_fp16.h>

typedef __attribute__((ext_vector_type(16))) _Float16 v16h;
typedef __attribute__((ext_vector_type(8)))  _Float16 v8h;
typedef __attribute__((ext_vector_type(8)))  float    v8f;

#define N_SEQ 4096
#define DIMH  32
#define NHEAD 4
#define BATCH 2
#define BH    (BATCH * NHEAD)   // 8
#define HID   128
#define CIN   64
// SCALE * log2(e): q is pre-multiplied so softmax uses a bare v_exp_f32 (exp2)
#define QKSCALE_LOG2E (10.0f * 1.4426950408889634f)

#define KB      64     // keys per staged tile
#define TILES   (N_SEQ / KB)
#define KROWSTR 40     // K-tile LDS row stride (halves): 32 data + 8 pad (bank-friendly)
#define VROWSTR 72     // V-tile LDS row stride (halves): 64 data + 8 pad
#define PROWSTR 72     // P-tile LDS row stride (halves)

// ---------------------------------------------------------------------------
// WMMA fragment helpers (layouts per CDNA5 ISA):
//  A 16x32 f16: lane l -> M=l%16, K = base + (e<8?e:e+8), base=(l<16?0:8)
//  B 32x16 f16: lane l -> N=l%16, K = base + (e<8?e:e+8), base=(l<16?0:8)
//  C/D 16x16 f32: lane l -> N=l%16, row M = v + 8*(l>=16)
// ---------------------------------------------------------------------------
__device__ __forceinline__ v16h load_frag_a(const _Float16* p) {
  v8h lo = *(const v8h*)p, hi8 = *(const v8h*)(p + 16);
  v16h r;
#pragma unroll
  for (int i = 0; i < 8; ++i) { r[i] = lo[i]; r[i + 8] = hi8[i]; }
  return r;
}
__device__ __forceinline__ v16h load_frag_b(const _Float16* p) {
  v8h lo = *(const v8h*)p, hi8 = *(const v8h*)(p + 16);
  v16h r;
#pragma unroll
  for (int i = 0; i < 8; ++i) { r[i] = lo[i]; r[i + 8] = hi8[i]; }
  return r;
}

// One wave32 instruction moves 512B (16B/lane) global -> LDS, tracked by ASYNCcnt.
__device__ __forceinline__ void async_cp16(unsigned lds_byte, unsigned gl_off,
                                           const void* sbase) {
  asm volatile("global_load_async_to_lds_b128 %0, %1, %2 offset:0"
               :: "v"(lds_byte), "v"(gl_off), "s"(sbase) : "memory");
}

// ---------------------------------------------------------------------------
// K1: qkv = w_qkv @ x.  x: [B][64][4096], w: [384][64], qkv: [B][384][4096]
// ---------------------------------------------------------------------------
__global__ void qkv_proj(const float* __restrict__ x, const float* __restrict__ w,
                         float* __restrict__ qkv) {
  int pos = blockIdx.x * 256 + threadIdx.x;
  int o   = blockIdx.y;
  int b   = blockIdx.z;
  const float* xb = x + (size_t)b * CIN * N_SEQ + pos;
  const float* wr = w + o * CIN;
  float acc = 0.f;
#pragma unroll 16
  for (int c = 0; c < CIN; ++c) acc = fmaf(wr[c], xb[(size_t)c * N_SEQ], acc);
  qkv[((size_t)b * 384 + o) * N_SEQ + pos] = acc;
}

// ---------------------------------------------------------------------------
// K2: per-(b,h,d) l2norm over N for q,k; fold QKSCALE*log2e into q; cast f16.
//   q,k -> [bh][N][32] row-major; v -> [bh][32][N]
// ---------------------------------------------------------------------------
__global__ void norm_pack(const float* __restrict__ qkv,
                          _Float16* __restrict__ qt, _Float16* __restrict__ kt,
                          _Float16* __restrict__ vt) {
  __shared__ float red[256];
  int r    = blockIdx.x;     // 0..767
  int type = r >> 8;         // 0=q 1=k 2=v
  int idx  = r & 255;
  int bh = idx >> 5, d = idx & 31;
  int b  = bh >> 2,  h = bh & 3;
  const float* src = qkv + ((size_t)b * 384 + type * HID + h * DIMH + d) * N_SEQ;
  int t = threadIdx.x;
  if (type == 2) {
    _Float16* dst = vt + ((size_t)bh * DIMH + d) * N_SEQ;
    for (int i = t; i < N_SEQ; i += 256) dst[i] = (_Float16)src[i];
    return;
  }
  float ss = 0.f;
  for (int i = t; i < N_SEQ; i += 256) { float v = src[i]; ss = fmaf(v, v, ss); }
  red[t] = ss; __syncthreads();
  for (int s = 128; s > 0; s >>= 1) { if (t < s) red[t] += red[t + s]; __syncthreads(); }
  float rn = 1.0f / fmaxf(sqrtf(red[0]), 1e-12f);
  if (type == 0) rn *= QKSCALE_LOG2E;
  _Float16* dst = (type == 0) ? qt : kt;
  for (int i = t; i < N_SEQ; i += 256)
    dst[((size_t)bh * N_SEQ + i) * DIMH + d] = (_Float16)(src[i] * rn);
}

// ---------------------------------------------------------------------------
// K3: flash attention, no-max softmax (exact here: l2norm bounds |sim*scale|
// << 1 for this data, softmax is shift-invariant, exp2 cannot overflow).
// Row sums computed by WMMA against a ones matrix -> zero cross-lane shuffles.
// 4 waves/block, 64 queries/block; 64-key K/V tiles double-buffered in LDS
// via global_load_async_to_lds_b128 (ASYNCcnt).
// ---------------------------------------------------------------------------
__global__ __launch_bounds__(128) void attn(const _Float16* __restrict__ qt,
                                            const _Float16* __restrict__ kt,
                                            const _Float16* __restrict__ vt,
                                            float* __restrict__ att) {
  __shared__ __align__(16) _Float16 kbuf[2][KB * KROWSTR];    //  2 x 5120 B
  __shared__ __align__(16) _Float16 vbuf[2][DIMH * VROWSTR];  //  2 x 4608 B
  __shared__ __align__(16) _Float16 pbuf[4][16 * PROWSTR];    //  4 x 2304 B

  int lane = threadIdx.x & 31;
  int wid  = threadIdx.x >> 5;
  int bh   = blockIdx.x >> 6;       // 8 (b,h) pairs
  int qblk = blockIdx.x & 63;       // 64 query blocks of 64
  int qbase = qblk * 64 + wid * 16; // this wave's 16 queries
  int nloc  = lane & 15;
  int hi    = lane >> 4;
  int koff  = hi ? 8 : 0;           // B-fragment K offset
  int abase = hi ? 8 : 0;           // A-fragment K base

  // Q A-fragment (reused across all key tiles)
  const _Float16* qrow = qt + ((size_t)bh * N_SEQ + qbase + nloc) * DIMH;
  v16h aq = load_frag_a(qrow + abase);

  v16h bones;                       // 32x16 ones matrix for row sums
#pragma unroll
  for (int i = 0; i < 16; ++i) bones[i] = (_Float16)1.0f;

  v8f o0 = {}, o1 = {}, lsum = {};

  const _Float16* ksrc0 = kt + (size_t)bh * N_SEQ * DIMH;  // [N][32], tiles contiguous
  const _Float16* vsrc0 = vt + (size_t)bh * DIMH * N_SEQ;  // [32][N]
  _Float16* pl = pbuf[wid];

  // cooperative async stage of one 64-key tile: 8 K-chunks + 8 V-chunks of
  // 512B each, 2+2 issued per wave.
  auto stage = [&](int t, int buf) {
    const _Float16* ks = ksrc0 + (size_t)t * KB * DIMH;
    const _Float16* vs = vsrc0 + (size_t)t * KB;
    unsigned kb = (unsigned)(uintptr_t)&kbuf[buf][0];
    unsigned vb = (unsigned)(uintptr_t)&vbuf[buf][0];
#pragma unroll
    for (int i = 0; i < 2; ++i) {
      int ci = wid * 2 + i;
      {
        int row = ci * 8 + (lane >> 2), ch = lane & 3;   // K rows of 64B
        async_cp16(kb + row * (KROWSTR * 2) + ch * 16, row * 64 + ch * 16, ks);
      }
      {
        int row = ci * 4 + (lane >> 3), ch = lane & 7;   // V rows of 128B, mem stride 8KB
        async_cp16(vb + row * (VROWSTR * 2) + ch * 16, row * (N_SEQ * 2) + ch * 16, vs);
      }
    }
  };

  stage(0, 0);
  for (int t = 0; t < TILES; ++t) {
    int cur = t & 1;
    if (t + 1 < TILES) {
      stage(t + 1, cur ^ 1);
      asm volatile("s_wait_asynccnt 0x4" ::: "memory");   // keep only the 4 new ones
    } else {
      asm volatile("s_wait_asynccnt 0x0" ::: "memory");
    }
    __syncthreads();                                       // tile t visible to all waves

    const _Float16* kc = kbuf[cur];
    const _Float16* vc = vbuf[cur];

    // S = q^T k for 4 chunks of 16 keys
    v8f s[4];
#pragma unroll
    for (int c = 0; c < 4; ++c) {
      v16h bk = load_frag_b(kc + (c * 16 + nloc) * KROWSTR + koff);
      v8f z = {};
      s[c] = __builtin_amdgcn_wmma_f32_16x16x32_f16(false, aq, false, bk, (short)0, z,
                                                    false, false);
    }
    // P = exp2(S) (scale*log2e pre-folded into q), store to per-wave LDS tile
#pragma unroll
    for (int c = 0; c < 4; ++c) {
      int col = c * 16 + nloc;
#pragma unroll
      for (int v = 0; v < 8; ++v) {
        int ml = v + (hi ? 8 : 0);
        pl[ml * PROWSTR + col] = (_Float16)__builtin_amdgcn_exp2f(s[c][v]);
      }
    }
    asm volatile("s_wait_dscnt 0x0" ::: "memory");
    v16h ap0 = load_frag_a(pl + nloc * PROWSTR + abase);        // keys 0..31
    v16h ap1 = load_frag_a(pl + nloc * PROWSTR + 32 + abase);   // keys 32..63

    // row sums via WMMA against ones (accumulated across all tiles)
    lsum = __builtin_amdgcn_wmma_f32_16x16x32_f16(false, ap0, false, bones, (short)0,
                                                  lsum, false, false);
    lsum = __builtin_amdgcn_wmma_f32_16x16x32_f16(false, ap1, false, bones, (short)0,
                                                  lsum, false, false);

    // O += P * V
#pragma unroll
    for (int kc2 = 0; kc2 < 2; ++kc2) {
      v16h apc = kc2 ? ap1 : ap0;
      v16h bv0 = load_frag_b(vc + nloc * VROWSTR + kc2 * 32 + koff);        // d 0..15
      v16h bv1 = load_frag_b(vc + (16 + nloc) * VROWSTR + kc2 * 32 + koff); // d 16..31
      o0 = __builtin_amdgcn_wmma_f32_16x16x32_f16(false, apc, false, bv0, (short)0, o0,
                                                  false, false);
      o1 = __builtin_amdgcn_wmma_f32_16x16x32_f16(false, apc, false, bv1, (short)0, o1,
                                                  false, false);
    }
    __syncthreads();   // all waves done with tile t before its buffer is restaged
  }

  // epilogue: divide by row sums (lsum replicated across columns in D layout)
#pragma unroll
  for (int v = 0; v < 8; ++v) { float r = 1.0f / lsum[v]; o0[v] *= r; o1[v] *= r; }
  size_t nb = (size_t)qbase + 8 * hi;   // 8 consecutive positions per lane, 32B aligned
  *(v8f*)(att + ((size_t)bh * DIMH + nloc) * N_SEQ + nb) = o0;
  *(v8f*)(att + ((size_t)bh * DIMH + 16 + nloc) * N_SEQ + nb) = o1;
}

// ---------------------------------------------------------------------------
// K4: out = w_out @ att + b_out.  att: [B][128][4096], out: [B][64][4096]
// ---------------------------------------------------------------------------
__global__ void out_proj(const float* __restrict__ att, const float* __restrict__ w,
                         const float* __restrict__ bias, float* __restrict__ out) {
  int pos = blockIdx.x * 256 + threadIdx.x;
  int o   = blockIdx.y;
  int b   = blockIdx.z;
  const float* ab = att + (size_t)b * HID * N_SEQ + pos;
  const float* wr = w + o * HID;
  float acc = bias[o];
#pragma unroll 16
  for (int c = 0; c < HID; ++c) acc = fmaf(wr[c], ab[(size_t)c * N_SEQ], acc);
  out[((size_t)b * 64 + o) * N_SEQ + pos] = acc;
}

// ---------------------------------------------------------------------------
extern "C" void kernel_launch(void* const* d_in, const int* in_sizes, int n_in,
                              void* d_out, int out_size, void* d_ws, size_t ws_size,
                              hipStream_t stream) {
  const float* x     = (const float*)d_in[0];
  const float* w_qkv = (const float*)d_in[1];
  const float* w_out = (const float*)d_in[2];
  const float* b_out = (const float*)d_in[3];
  float* out = (float*)d_out;

  char* ws = (char*)d_ws;
  size_t off = 0;
  auto carve = [&](size_t bytes) -> void* {
    void* p = ws + off;
    off = (off + bytes + 255) & ~(size_t)255;
    return p;
  };
  float*    qkv = (float*)   carve(sizeof(float)    * BATCH * 384 * N_SEQ);
  _Float16* qtp = (_Float16*)carve(sizeof(_Float16) * BH * N_SEQ * DIMH);
  _Float16* ktp = (_Float16*)carve(sizeof(_Float16) * BH * N_SEQ * DIMH);
  _Float16* vtp = (_Float16*)carve(sizeof(_Float16) * BH * DIMH * N_SEQ);
  float*    att = (float*)   carve(sizeof(float)    * BATCH * HID * N_SEQ);
  (void)ws_size; (void)in_sizes; (void)n_in; (void)out_size;

  qkv_proj<<<dim3(N_SEQ / 256, 3 * HID, BATCH), 256, 0, stream>>>(x, w_qkv, qkv);
  norm_pack<<<dim3(3 * 256), 256, 0, stream>>>(qkv, qtp, ktp, vtp);
  attn<<<dim3(BH * 64), 128, 0, stream>>>(qtp, ktp, vtp, att);
  out_proj<<<dim3(N_SEQ / 256, 64, BATCH), 256, 0, stream>>>(att, w_out, b_out, out);
}
